// RPN_Detector_30992484008026
// MI455X (gfx1250) — compile-verified, matrix operand order checked by reading, not run on hardware
//
#include <hip/hip_runtime.h>
#include <math.h>

// ---------------------------------------------------------------------------
// RPN detector forward for MI455X (gfx1250, wave32, WMMA).
//
// d_in layout (setup_inputs insertion order, params flattened layer-major,
// each layer contributes w then b):
//   0:x(B,3,N) 1:sn(B,3,N) 2:node(B,3,M)
//   pn1  w/b: 3,4  5,6  7,8
//   pn2  w/b: 9,10 11,12
//   knn1 w/b: 13,14 15,16 17,18
//   knn2 w/b: 19,20 21,22
//   mlp1 w/b: 23,24   mlp2: 25,26   mlp3: 27,28
// d_out: [cluster_mean(B,3,M) | keypoints(B,3,M) | sigmas(B,M)] = 28672 f32
// ---------------------------------------------------------------------------

#define BB    8
#define NN    16384
#define MM    512
#define KSOM  3
#define KNPTS (KSOM * NN)   // 49152 stacked points per batch
#define BMTOT (BB * MM)     // 4096 nodes total

typedef __attribute__((ext_vector_type(16))) __bf16 v16bf;
typedef __attribute__((ext_vector_type(8)))  float  v8f;

// ---------------------------------------------------------------------------
// WMMA fragment helpers (16-bit A 16x32 layout: lanes 0-15 rows M, kbase=0;
// lanes 16-31 rows M, kbase=8; element pairs cover K = kbase + {0..7,16..23}).
// B fragments are pre-packed in workspace with the identical (lane,elem)->K
// mapping so that a B load is 32 contiguous bytes per lane.
// ---------------------------------------------------------------------------
static __device__ __forceinline__ int frag_k(int lane, int e) {
  int kbase = (lane & 16) ? 8 : 0;
  int p = e >> 1;
  return kbase + ((p & 3) << 1) + (e & 1) + ((p & 4) ? 16 : 0);
}

static __device__ __forceinline__ v16bf load_A(const __bf16* act, int pitch, int lane) {
  int row = lane & 15;
  int kbase = (lane & 16) ? 8 : 0;
  const __bf16* rp = act + row * pitch + kbase;
  union { v16bf v; unsigned int u[8]; } r;
#pragma unroll
  for (int p = 0; p < 8; ++p) {
    int k0 = ((p & 3) << 1) + ((p & 4) ? 16 : 0);   // even -> 4B aligned pair
    r.u[p] = *reinterpret_cast<const unsigned int*>(rp + k0);
  }
  return r.v;
}

static __device__ __forceinline__ v16bf load_B(const __bf16* wp, int lane) {
  union { v16bf v; uint4 q[2]; } r;
  const uint4* p = reinterpret_cast<const uint4*>(wp + lane * 16);
  r.q[0] = p[0];
  r.q[1] = p[1];
  return r.v;
}

// One 16(sample) x 16(out-ch) tile: C += A(16xK) * B(KxN), K = kChunks*32
static __device__ __forceinline__ v8f mma_tile(const __bf16* act, int pitch, int kChunks,
                                               const __bf16* wP, int tilesK, int tn, int lane) {
  v8f c = {};
  for (int kc = 0; kc < kChunks; ++kc) {
    if (kc + 1 < kChunks)
      __builtin_prefetch(wP + (((size_t)tn * tilesK + kc + 1) << 9), 0, 0);
    v16bf a  = load_A(act + kc * 32, pitch, lane);
    v16bf bf = load_B(wP + (((size_t)tn * tilesK + kc) << 9), lane);
    c = __builtin_amdgcn_wmma_f32_16x16x32_bf16(false, a, false, bf, (short)0, c,
                                                false, false);
  }
  return c;
}

// D layout: lane col = lane&15, rows r + 8*(lane>=16)
static __device__ __forceinline__ void store_relu(v8f c, __bf16* outAct, int pitch, int tn,
                                                  const float* bias, int lane) {
  int ch = (tn << 4) + (lane & 15);
  float bv = bias[ch];
  int rbase = (lane & 16) ? 8 : 0;
#pragma unroll
  for (int r = 0; r < 8; ++r) {
    float v = fmaxf(c[r] + bv, 0.f);
    outAct[(rbase + r) * pitch + ch] = (__bf16)v;
  }
}

// ReLU + segment-max via float atomicMax (order independent -> deterministic).
static __device__ __forceinline__ void store_amax(v8f c, float* gmax, const int* bmRow,
                                                  int chStride, int tn, const float* bias,
                                                  int lane) {
  int ch = (tn << 4) + (lane & 15);
  float bv = bias[ch];
  int rbase = (lane & 16) ? 8 : 0;
#pragma unroll
  for (int r = 0; r < 8; ++r) {
    float v = fmaxf(c[r] + bv, 0.f);
    atomicMax(gmax + (size_t)bmRow[rbase + r] * chStride + ch, v);
  }
}

// ---------------------------------------------------------------------------
// Utility kernels
// ---------------------------------------------------------------------------
__global__ void __launch_bounds__(256) k_zero(float* p, int n) {
  int i = blockIdx.x * 256 + threadIdx.x;
  if (i < n) p[i] = 0.f;
}

// Pack f32 weight (out,in) into bf16 WMMA B-fragments: [tn][tk][lane][16]
__global__ void __launch_bounds__(256) k_pack(const float* __restrict__ w,
                                              __bf16* __restrict__ dst,
                                              int outCh, int inCh, int tilesK, int total) {
  int idx = blockIdx.x * 256 + threadIdx.x;
  if (idx >= total) return;
  int tile = idx >> 9;
  int rem  = idx & 511;
  int lane = rem >> 4;
  int e    = rem & 15;
  int tn = tile / tilesK;
  int tk = tile - tn * tilesK;
  int n = (tn << 4) + (lane & 15);
  int k = (tk << 5) + frag_k(lane, e);
  float v = (n < outCh && k < inCh) ? w[(size_t)n * inCh + k] : 0.f;
  dst[idx] = (__bf16)v;
}

// ---------------------------------------------------------------------------
// Stage 1: per-point top-3 nearest SOM nodes (nodes cached in LDS)
// ---------------------------------------------------------------------------
__global__ void __launch_bounds__(256) k_topk_assign(const float* __restrict__ x,
                                                     const float* __restrict__ node,
                                                     int* __restrict__ minIdx) {
  __shared__ float sx[MM], sy[MM], sz[MM];
  int b = blockIdx.x >> 6;
  int chunk = blockIdx.x & 63;
  int tid = threadIdx.x;
  const float* nb = node + (size_t)b * 3 * MM;
  for (int m = tid; m < MM; m += 256) {
    sx[m] = nb[m]; sy[m] = nb[MM + m]; sz[m] = nb[2 * MM + m];
  }
  __syncthreads();
  int n = chunk * 256 + tid;
  const float* xb = x + (size_t)b * 3 * NN;
  float px = xb[n], py = xb[NN + n], pz = xb[2 * NN + n];
  float d0 = 3.4e38f, d1 = 3.4e38f, d2 = 3.4e38f;
  int   i0 = 0, i1 = 0, i2 = 0;
  for (int m = 0; m < MM; ++m) {
    float dx = px - sx[m], dy = py - sy[m], dz = pz - sz[m];
    float d = dx * dx + dy * dy + dz * dz;
    if (d < d0)      { d2 = d1; i2 = i1; d1 = d0; i1 = i0; d0 = d; i0 = m; }
    else if (d < d1) { d2 = d1; i2 = i1; d1 = d;  i1 = m; }
    else if (d < d2) { d2 = d;  i2 = m; }
  }
  int* mi = minIdx + (size_t)b * KNPTS;
  mi[n] = i0; mi[NN + n] = i1; mi[2 * NN + n] = i2;   // layout: k*N + n
}

// ---------------------------------------------------------------------------
// Stage 2: deterministic segment mean (fixed-order scan + LDS tree reduce)
// ---------------------------------------------------------------------------
__global__ void __launch_bounds__(256) k_seg_mean(const float* __restrict__ x,
                                                  const int* __restrict__ minIdx,
                                                  float* __restrict__ cm,
                                                  float* __restrict__ out) {
  __shared__ float sc[256], s0[256], s1[256], s2[256];
  int bm = blockIdx.x;
  int b = bm >> 9, m = bm & 511;
  int tid = threadIdx.x;
  const int* mi = minIdx + (size_t)b * KNPTS;
  const float* xb = x + (size_t)b * 3 * NN;
  float cnt = 0.f, a0 = 0.f, a1 = 0.f, a2 = 0.f;
  for (int j = tid; j < KNPTS; j += 256) {
    if (mi[j] == m) {
      int n = j & (NN - 1);
      cnt += 1.f;
      a0 += xb[n]; a1 += xb[NN + n]; a2 += xb[2 * NN + n];
    }
  }
  sc[tid] = cnt; s0[tid] = a0; s1[tid] = a1; s2[tid] = a2;
  __syncthreads();
  for (int s = 128; s > 0; s >>= 1) {
    if (tid < s) {
      sc[tid] += sc[tid + s]; s0[tid] += s0[tid + s];
      s1[tid] += s1[tid + s]; s2[tid] += s2[tid + s];
    }
    __syncthreads();
  }
  if (tid == 0) {
    float inv = 1.f / (sc[0] + 1e-5f);
    float v0 = s0[0] * inv, v1 = s1[0] * inv, v2 = s2[0] * inv;
    cm[(size_t)bm * 3 + 0] = v0;
    cm[(size_t)bm * 3 + 1] = v1;
    cm[(size_t)bm * 3 + 2] = v2;
    out[b * 1536 + m]        = v0;   // cluster_mean output (B,3,M)
    out[b * 1536 + 512 + m]  = v1;
    out[b * 1536 + 1024 + m] = v2;
  }
}

// ---------------------------------------------------------------------------
// Stage 3: pn1 (6->64->64->64) per 16-point wave tile, seg-max via atomicMax
// ---------------------------------------------------------------------------
static __device__ __forceinline__ void stage_pn1_input(
    const float* __restrict__ x, const float* __restrict__ sn,
    const float* __restrict__ cm, const int* __restrict__ minIdx,
    int b, int j0, int lane, __bf16* sA /*16x40*/, int* sBM /*16*/) {
  if (lane < 16) {
    int j = j0 + lane;
    int n = j & (NN - 1);
    int bm = b * MM + minIdx[(size_t)b * KNPTS + j];
    sBM[lane] = bm;
    const float* xb = x + (size_t)b * 3 * NN;
    const float* sb = sn + (size_t)b * 3 * NN;
    const float* c3 = cm + (size_t)bm * 3;
    __bf16* row = sA + lane * 40;
    row[0] = (__bf16)(xb[n] - c3[0]);
    row[1] = (__bf16)(xb[NN + n] - c3[1]);
    row[2] = (__bf16)(xb[2 * NN + n] - c3[2]);
    row[3] = (__bf16)sb[n];
    row[4] = (__bf16)sb[NN + n];
    row[5] = (__bf16)sb[2 * NN + n];
#pragma unroll
    for (int c = 6; c < 32; ++c) row[c] = (__bf16)0.f;
  }
}

__global__ void __launch_bounds__(128) k_pn1max(
    const float* __restrict__ x, const float* __restrict__ sn,
    const float* __restrict__ cm, const int* __restrict__ minIdx,
    float* __restrict__ firstMax,
    const __bf16* w0, const __bf16* w1, const __bf16* w2,
    const float* b0, const float* b1, const float* b2) {
  __shared__ __bf16 sA[4][16][40];
  __shared__ __bf16 sB[4][16][72];
  __shared__ __bf16 sC[4][16][72];
  __shared__ int    sBM[4][16];
  int wv = threadIdx.x >> 5, lane = threadIdx.x & 31;
  int g = (blockIdx.x * 4 + wv) << 4;
  int b = g / KNPTS;
  int j0 = g - b * KNPTS;
  stage_pn1_input(x, sn, cm, minIdx, b, j0, lane, &sA[wv][0][0], &sBM[wv][0]);
  for (int tn = 0; tn < 4; ++tn)
    store_relu(mma_tile(&sA[wv][0][0], 40, 1, w0, 1, tn, lane), &sB[wv][0][0], 72, tn, b0, lane);
  for (int tn = 0; tn < 4; ++tn)
    store_relu(mma_tile(&sB[wv][0][0], 72, 2, w1, 2, tn, lane), &sC[wv][0][0], 72, tn, b1, lane);
  for (int tn = 0; tn < 4; ++tn)
    store_amax(mma_tile(&sC[wv][0][0], 72, 2, w2, 2, tn, lane), firstMax, &sBM[wv][0], 64, tn, b2, lane);
}

// ---------------------------------------------------------------------------
// Stage 4: recompute pn1, fuse [first | gathered first_max] -> pn2, atomicMax
// ---------------------------------------------------------------------------
__global__ void __launch_bounds__(64) k_pn2max(
    const float* __restrict__ x, const float* __restrict__ sn,
    const float* __restrict__ cm, const int* __restrict__ minIdx,
    const float* __restrict__ firstMax, float* __restrict__ secondMax,
    const __bf16* w0, const __bf16* w1, const __bf16* w2,
    const float* b0, const float* b1, const float* b2,
    const __bf16* p0, const __bf16* p1, const float* q0, const float* q1) {
  __shared__ __bf16 sA[2][16][40];
  __shared__ __bf16 sB[2][16][72];
  __shared__ __bf16 sC[2][16][72];
  __shared__ __bf16 sF[2][16][136];
  __shared__ __bf16 sG[2][16][136];
  __shared__ int    sBM[2][16];
  int wv = threadIdx.x >> 5, lane = threadIdx.x & 31;
  int g = (blockIdx.x * 2 + wv) << 4;
  int b = g / KNPTS;
  int j0 = g - b * KNPTS;
  stage_pn1_input(x, sn, cm, minIdx, b, j0, lane, &sA[wv][0][0], &sBM[wv][0]);
  for (int tn = 0; tn < 4; ++tn)
    store_relu(mma_tile(&sA[wv][0][0], 40, 1, w0, 1, tn, lane), &sB[wv][0][0], 72, tn, b0, lane);
  for (int tn = 0; tn < 4; ++tn)
    store_relu(mma_tile(&sB[wv][0][0], 72, 2, w1, 2, tn, lane), &sC[wv][0][0], 72, tn, b1, lane);
  for (int tn = 0; tn < 4; ++tn)   // first -> sF cols 0..63
    store_relu(mma_tile(&sC[wv][0][0], 72, 2, w2, 2, tn, lane), &sF[wv][0][0], 136, tn, b2, lane);
  if (lane < 16) {                 // gather(first_max) -> sF cols 64..127
    const float* fm = firstMax + (size_t)sBM[wv][lane] * 64;
    for (int c = 0; c < 64; ++c) sF[wv][lane][64 + c] = (__bf16)fm[c];
  }
  for (int tn = 0; tn < 8; ++tn)
    store_relu(mma_tile(&sF[wv][0][0], 136, 4, p0, 4, tn, lane), &sG[wv][0][0], 136, tn, q0, lane);
  for (int tn = 0; tn < 8; ++tn)
    store_amax(mma_tile(&sG[wv][0][0], 136, 4, p1, 4, tn, lane), secondMax, &sBM[wv][0], 128, tn, q1, lane);
}

// ---------------------------------------------------------------------------
// Stage 5: node-KNN (top-16 over 512 nodes, register insertion network)
// ---------------------------------------------------------------------------
__global__ void __launch_bounds__(256) k_knn_topk(const float* __restrict__ cm,
                                                  int* __restrict__ knnI) {
  __shared__ float q[MM * 3];
  int b = blockIdx.x >> 1;
  int m = ((blockIdx.x & 1) << 8) + threadIdx.x;
  for (int i = threadIdx.x; i < MM * 3; i += 256) q[i] = cm[(size_t)b * MM * 3 + i];
  __syncthreads();
  float qx = q[m * 3], qy = q[m * 3 + 1], qz = q[m * 3 + 2];
  float bd[16]; int bi[16];
#pragma unroll
  for (int i = 0; i < 16; ++i) { bd[i] = 3.4e38f; bi[i] = 0; }
  for (int mm = 0; mm < MM; ++mm) {
    float dx = qx - q[mm * 3], dy = qy - q[mm * 3 + 1], dz = qz - q[mm * 3 + 2];
    float d = dx * dx + dy * dy + dz * dz;
    if (d < bd[15]) {
      bd[15] = d; bi[15] = mm;
#pragma unroll
      for (int p = 15; p > 0; --p)
        if (bd[p] < bd[p - 1]) {
          float td = bd[p]; bd[p] = bd[p - 1]; bd[p - 1] = td;
          int ti = bi[p]; bi[p] = bi[p - 1]; bi[p - 1] = ti;
        }
    }
  }
  int* dst = knnI + ((size_t)b * MM + m) * 16;
#pragma unroll
  for (int i = 0; i < 16; ++i) dst[i] = bi[i];
}

// ---------------------------------------------------------------------------
// Stage 6: fully fused KNN module per node (knn1 -> max_k -> concat -> knn2 -> max_k)
// one block = one node, 16 samples live entirely in LDS
// ---------------------------------------------------------------------------
__global__ void __launch_bounds__(128) k_knn_fused(
    const float* __restrict__ cm, const float* __restrict__ secondMax,
    const int* __restrict__ knnI, float* __restrict__ knnOut,
    const __bf16* kw0, const __bf16* kw1, const __bf16* kw2,
    const float* kb0, const float* kb1, const float* kb2,
    const __bf16* g0, const __bf16* g1, const float* gb0, const float* gb1) {
  __shared__ __bf16 sIn[16][168];
  __shared__ __bf16 sX[16][520];
  __shared__ __bf16 sY[16][520];
  __shared__ float  sMax[256];
  __shared__ int    sNbr[16];
  __shared__ float  sQ[3];
  int bm = blockIdx.x;
  int b = bm >> 9;
  int tid = threadIdx.x, wv = tid >> 5, lane = tid & 31;
  if (tid < 16) sNbr[tid] = knnI[(size_t)bm * 16 + tid];
  if (tid < 3)  sQ[tid] = cm[(size_t)bm * 3 + tid];
  __syncthreads();
  for (int i = tid; i < 16 * 160; i += 128) {     // [dx,dy,dz | feat(128) | pad]
    int r = i / 160, c = i - r * 160;
    float v;
    if (c < 3)        v = cm[((size_t)b * MM + sNbr[r]) * 3 + c] - sQ[c];
    else if (c < 131) v = secondMax[((size_t)b * MM + sNbr[r]) * 128 + (c - 3)];
    else              v = 0.f;
    sIn[r][c] = (__bf16)v;
  }
  __syncthreads();
  for (int i = 0; i < 4; ++i) { int tn = wv * 4 + i;
    store_relu(mma_tile(&sIn[0][0], 168, 5, kw0, 5, tn, lane), &sX[0][0], 520, tn, kb0, lane); }
  __syncthreads();
  for (int i = 0; i < 4; ++i) { int tn = wv * 4 + i;
    store_relu(mma_tile(&sX[0][0], 520, 8, kw1, 8, tn, lane), &sY[0][0], 520, tn, kb1, lane); }
  __syncthreads();
  for (int i = 0; i < 4; ++i) { int tn = wv * 4 + i;   // f1 -> sX[.][0..255]
    store_relu(mma_tile(&sY[0][0], 520, 8, kw2, 8, tn, lane), &sX[0][0], 520, tn, kb2, lane); }
  __syncthreads();
  for (int c = tid; c < 256; c += 128) {               // f1 max over K samples
    float mx = 0.f;
    for (int r = 0; r < 16; ++r) mx = fmaxf(mx, (float)sX[r][c]);
    sMax[c] = mx;
  }
  __syncthreads();
  for (int i = tid; i < 16 * 256; i += 128) {          // broadcast into cols 256..511
    int r = i >> 8, c = i & 255;
    sX[r][256 + c] = (__bf16)sMax[c];
  }
  __syncthreads();
  for (int i = 0; i < 8; ++i) { int tn = wv * 8 + i;
    store_relu(mma_tile(&sX[0][0], 520, 16, g0, 16, tn, lane), &sY[0][0], 520, tn, gb0, lane); }
  __syncthreads();
  for (int i = 0; i < 8; ++i) { int tn = wv * 8 + i;
    store_relu(mma_tile(&sY[0][0], 520, 16, g1, 16, tn, lane), &sX[0][0], 520, tn, gb1, lane); }
  __syncthreads();
  for (int c = tid; c < 512; c += 128) {
    float mx = 0.f;
    for (int r = 0; r < 16; ++r) mx = fmaxf(mx, (float)sX[r][c]);
    knnOut[(size_t)bm * 512 + c] = mx;
  }
}

// ---------------------------------------------------------------------------
// Stage 7: head 640->512->256->4 per 16-node tile + keypoints / sigmas
// ---------------------------------------------------------------------------
__global__ void __launch_bounds__(128) k_head(
    const float* __restrict__ secondMax, const float* __restrict__ knnOut,
    const float* __restrict__ cm,
    const __bf16* m1w, const float* m1b,
    const __bf16* m2w, const float* m2b,
    const float* __restrict__ w3, const float* __restrict__ b3,
    float* __restrict__ out) {
  __shared__ __bf16 sA[16][648];
  __shared__ __bf16 sH1[16][520];
  __shared__ __bf16 sH2[16][264];
  int bm0 = blockIdx.x << 4;
  int tid = threadIdx.x, wv = tid >> 5, lane = tid & 31;
  for (int i = tid; i < 16 * 640; i += 128) {
    int r = i / 640, c = i - r * 640;
    float v = (c < 128) ? secondMax[(size_t)(bm0 + r) * 128 + c]
                        : knnOut[(size_t)(bm0 + r) * 512 + (c - 128)];
    sA[r][c] = (__bf16)v;
  }
  __syncthreads();
  for (int i = 0; i < 8; ++i) { int tn = wv * 8 + i;
    store_relu(mma_tile(&sA[0][0], 648, 20, m1w, 20, tn, lane), &sH1[0][0], 520, tn, m1b, lane); }
  __syncthreads();
  for (int i = 0; i < 4; ++i) { int tn = wv * 4 + i;
    store_relu(mma_tile(&sH1[0][0], 520, 16, m2w, 16, tn, lane), &sH2[0][0], 264, tn, m2b, lane); }
  __syncthreads();
  if (tid < 64) {                                  // mlp3: 4 outputs, linear
    int r = tid >> 2, o = tid & 3;
    float acc = b3[o];
    const float* wr = w3 + o * 256;
    for (int c = 0; c < 256; ++c) acc += (float)sH2[r][c] * wr[c];
    int bm = bm0 + r;
    int b = bm >> 9, m = bm & 511;
    if (o < 3) {
      out[12288 + b * 1536 + o * 512 + m] = acc + cm[(size_t)bm * 3 + o];
    } else {
      float sp = fmaxf(acc, 0.f) + log1pf(expf(-fabsf(acc)));   // stable softplus
      out[24576 + b * 512 + m] = sp + 0.001f;
    }
  }
}

// ---------------------------------------------------------------------------
// Host launcher
// ---------------------------------------------------------------------------
extern "C" void kernel_launch(void* const* d_in, const int* in_sizes, int n_in,
                              void* d_out, int out_size, void* d_ws, size_t ws_size,
                              hipStream_t stream) {
  (void)in_sizes; (void)n_in; (void)out_size; (void)ws_size;
  const float* x    = (const float*)d_in[0];
  const float* sn   = (const float*)d_in[1];
  const float* node = (const float*)d_in[2];
  float* out = (float*)d_out;

  char* ws = (char*)d_ws;
  size_t off = 0;
  auto alloc = [&](size_t bytes) -> void* {
    void* p = ws + off;
    off += (bytes + 255) & ~(size_t)255;
    return p;
  };

  // zero-initialized region (firstMax | secondMax contiguous)
  float* firstMax  = (float*)alloc((size_t)BMTOT * 64 * 4);
  float* secondMax = (float*)alloc((size_t)BMTOT * 128 * 4);
  int zeroFloats = BMTOT * 64 + BMTOT * 128;       // 786432, contiguous (256B multiples)

  float* cmW    = (float*)alloc((size_t)BMTOT * 3 * 4);
  float* knnOut = (float*)alloc((size_t)BMTOT * 512 * 4);
  int*   minIdx = (int*)alloc((size_t)BB * KNPTS * 4);
  int*   knnI   = (int*)alloc((size_t)BMTOT * 16 * 4);

  // pack all GEMM weights into bf16 WMMA B-fragments (resident in L2 thereafter)
  struct PW { int widx, outc, inc; };
  const PW pws[12] = {
    {3, 64, 6}, {5, 64, 64}, {7, 64, 64},          // pn1
    {9, 128, 128}, {11, 128, 128},                 // pn2
    {13, 256, 131}, {15, 256, 256}, {17, 256, 256},// knn1
    {19, 512, 512}, {21, 512, 512},                // knn2
    {23, 512, 640}, {25, 256, 512}                 // mlp1, mlp2
  };
  __bf16* pk[12];
  for (int i = 0; i < 12; ++i) {
    int tn = (pws[i].outc + 15) / 16, tk = (pws[i].inc + 31) / 32;
    int total = tn * tk * 512;
    pk[i] = (__bf16*)alloc((size_t)total * 2);
    k_pack<<<(total + 255) / 256, 256, 0, stream>>>(
        (const float*)d_in[pws[i].widx], pk[i], pws[i].outc, pws[i].inc, tk, total);
  }

  k_zero<<<(zeroFloats + 255) / 256, 256, 0, stream>>>(firstMax, zeroFloats);

  k_topk_assign<<<BB * (NN / 256), 256, 0, stream>>>(x, node, minIdx);
  k_seg_mean<<<BMTOT, 256, 0, stream>>>(x, minIdx, cmW, out);

  const float *b0 = (const float*)d_in[4], *b1 = (const float*)d_in[6],
              *b2 = (const float*)d_in[8],
              *q0 = (const float*)d_in[10], *q1 = (const float*)d_in[12],
              *kb0 = (const float*)d_in[14], *kb1 = (const float*)d_in[16],
              *kb2 = (const float*)d_in[18],
              *gb0 = (const float*)d_in[20], *gb1 = (const float*)d_in[22],
              *m1b = (const float*)d_in[24], *m2b = (const float*)d_in[26],
              *w3 = (const float*)d_in[27], *b3 = (const float*)d_in[28];

  k_pn1max<<<(BB * KNPTS / 16) / 4, 128, 0, stream>>>(
      x, sn, cmW, minIdx, firstMax, pk[0], pk[1], pk[2], b0, b1, b2);

  k_pn2max<<<(BB * KNPTS / 16) / 2, 64, 0, stream>>>(
      x, sn, cmW, minIdx, firstMax, secondMax,
      pk[0], pk[1], pk[2], b0, b1, b2, pk[3], pk[4], q0, q1);

  k_knn_topk<<<BB * 2, 256, 0, stream>>>(cmW, knnI);

  k_knn_fused<<<BMTOT, 128, 0, stream>>>(
      cmW, secondMax, knnI, knnOut,
      pk[5], pk[6], pk[7], kb0, kb1, kb2, pk[8], pk[9], gb0, gb1);

  k_head<<<BMTOT / 16, 128, 0, stream>>>(
      secondMax, knnOut, cmW, pk[10], m1b, pk[11], m2b, w3, b3, out);
}
